// SelfAttention_66159676228012
// MI455X (gfx1250) — compile-verified, matrix operand order checked by reading, not run on hardware
//
#include <hip/hip_runtime.h>
#include <hip/hip_bf16.h>

// ---------------------------------------------------------------------------
// Self-attention B=4, S=2048, D=1024 on gfx1250 (MI455X).
// GEMMs on v_wmma_f32_16x16x32_bf16; tile staging via CDNA5 TDM
// (tensor_load_to_lds, TENSORcnt) and async global->LDS (ASYNCcnt),
// double-buffered.
// ws layout (bytes):
//   [0, 16.78M)      Q  bf16 [4*2048, 1024]
//   [16.78M, 33.55M) K  bf16 [4*2048, 1024]
//   [33.55M, 50.33M) V^T bf16 [4][1024][2048]   (transposed at projection)
//   [50.33M, 117.4M) scores f32 [4,2048,2048]; softmax rewrites rows in
//                    place as bf16 probabilities (row byte-stride 8192).
// ---------------------------------------------------------------------------

#define DMODEL 1024
#define SEQ    2048
#define NBATCH 4
#define MTOT   (NBATCH * SEQ)

typedef __attribute__((ext_vector_type(16))) __bf16 v16bf;
typedef __attribute__((ext_vector_type(8)))  __bf16 bf16x8;
typedef __attribute__((ext_vector_type(8)))  float  v8f;
typedef __attribute__((ext_vector_type(4)))  float  f32x4;
typedef __attribute__((ext_vector_type(8)))  float  f32x8;
typedef __attribute__((ext_vector_type(4)))  unsigned u32x4;
typedef __attribute__((ext_vector_type(8)))  unsigned u32x8;
typedef __attribute__((ext_vector_type(4)))  int      i32x4;
typedef __attribute__((ext_vector_type(8)))  int      i32x8;

// ---- WMMA -------------------------------------------------------------------
__device__ __forceinline__ v8f wmma_bf16(v16bf a, v16bf b, v8f c) {
  return __builtin_amdgcn_wmma_f32_16x16x32_bf16(false, a, false, b,
                                                 (short)0, c, false, false);
}

// ---- LDS byte address (low 32 bits of generic pointer = AS(3) offset) ------
__device__ __forceinline__ unsigned lds_off(const void* p) {
  return (unsigned)(unsigned long long)p;
}

// ---- async global -> LDS, 16B per lane (ASYNCcnt) ---------------------------
__device__ __forceinline__ void async_copy_b128(unsigned lds_addr, const void* g) {
  asm volatile("global_load_async_to_lds_b128 %0, %1, off"
               :: "v"(lds_addr), "v"((unsigned long long)g)
               : "memory");
}
__device__ __forceinline__ void wait_async0() {
#if __has_builtin(__builtin_amdgcn_s_wait_asynccnt)
  __builtin_amdgcn_s_wait_asynccnt(0);
#else
  asm volatile("s_wait_asynccnt 0x0" ::: "memory");
#endif
}

// ---- TDM: 2-D tile (tile0 x tile1, 2-byte elems) global -> LDS --------------
__device__ __forceinline__ void tdm_load_2d_bf16(unsigned lds_addr, const void* g,
                                                 unsigned tile0, unsigned tile1,
                                                 unsigned long long stride0,
                                                 unsigned tdim0, unsigned tdim1) {
  unsigned long long ga = (unsigned long long)g;
  u32x4 g0;
  g0[0] = 1u;                                   // count=1, user descriptor
  g0[1] = lds_addr;                             // LDS byte address
  g0[2] = (unsigned)ga;                         // global_addr[31:0]
  g0[3] = (unsigned)((ga >> 32) & 0x01FFFFFFu) | 0x80000000u; // addr[56:32]|type=2
  u32x8 g1;
  g1[0] = 1u << 16;                             // data_size=1 (2 bytes)
  g1[1] = (tdim0 & 0xFFFFu) << 16;              // tensor_dim0[15:0]
  g1[2] = (tdim0 >> 16) | ((tdim1 & 0xFFFFu) << 16);
  g1[3] = (tdim1 >> 16) | (tile0 << 16);        // tile_dim0
  g1[4] = tile1;                                // tile_dim1 (tile_dim2=0)
  g1[5] = (unsigned)stride0;                    // tensor_dim0_stride[31:0]
  g1[6] = (unsigned)(stride0 >> 32);            // stride0[47:32] | stride1=0
  g1[7] = 0u;
#if __has_builtin(__builtin_amdgcn_tensor_load_to_lds)
  i32x4 z4 = {0, 0, 0, 0};
  i32x8 z8 = {0, 0, 0, 0, 0, 0, 0, 0};
  __builtin_amdgcn_tensor_load_to_lds(g0, __builtin_bit_cast(i32x8, g1),
                                      z4, z4, z8, 0);
#else
  asm volatile("tensor_load_to_lds %0, %1" :: "s"(g0), "s"(g1) : "memory");
#endif
}
__device__ __forceinline__ void wait_tensor0() {
  __builtin_amdgcn_s_wait_tensorcnt(0);
}

// ---- fragment loaders (wave32 16-bit A/B layout) ----------------------------
// lane l -> row/col l&15; lanes 0-15: K={0..7,16..23}, lanes 16-31: {8..15,24..31}
__device__ __forceinline__ v16bf frag_bf16(const __bf16* rowptr, int koff) {
  bf16x8 lo = *reinterpret_cast<const bf16x8*>(rowptr + koff);
  bf16x8 hi = *reinterpret_cast<const bf16x8*>(rowptr + koff + 16);
  return __builtin_shufflevector(lo, hi, 0,1,2,3,4,5,6,7,8,9,10,11,12,13,14,15);
}
__device__ __forceinline__ v16bf frag_f32(const float* rowptr, int koff) {
  f32x8 lo = *reinterpret_cast<const f32x8*>(rowptr + koff);
  f32x8 hi = *reinterpret_cast<const f32x8*>(rowptr + koff + 16);
  v16bf r;
#pragma unroll
  for (int i = 0; i < 8; ++i) { r[i] = (__bf16)lo[i]; r[i + 8] = (__bf16)hi[i]; }
  return r;
}

// ---------------------------------------------------------------------------
// Kernel 1: P = X*W^T + bias (M=8192,N=1024,K=1024), bf16 out.
// W f32 tile (64x32) staged via async global->LDS b128, double-buffered;
// converted to bf16 at fragment build. TRANS=true writes V^T with packed
// 16B stores (WMMA C layout: 8 consecutive rows per lane at fixed column).
// ---------------------------------------------------------------------------
template <bool TRANS>
__global__ __launch_bounds__(256)
void proj_kernel(const float* __restrict__ X, const float* __restrict__ W,
                 const float* __restrict__ bias, __bf16* __restrict__ Out) {
  __shared__ float ldsW[2][64 * 32];

  const int tid   = threadIdx.x;
  const int wave  = tid >> 5;
  const int lane  = tid & 31;
  const int l16   = lane & 15;
  const int lhalf = lane >> 4;
  const int klo   = lhalf * 8;

  const int mtile = blockIdx.x * 128;
  const int ntile = blockIdx.y * 64;

  const float* xrow = X + (size_t)(mtile + wave * 16 + l16) * DMODEL;

  const int r0 = tid >> 3;                 // staging: row, 16B column chunk
  const int c0 = (tid & 7) << 2;
  const float* wsrc = W + (size_t)(ntile + r0) * DMODEL + c0;

  v8f acc[4] = {};
  const int NT = DMODEL / 32;

  // prologue: stage tile 0 into buffer 0
#pragma unroll
  for (int h = 0; h < 2; ++h)
    async_copy_b128(lds_off(&ldsW[0][(r0 + 32 * h) * 32 + c0]),
                    wsrc + (size_t)(32 * h) * DMODEL);
  wait_async0();
  __syncthreads();

  for (int i = 0; i < NT; ++i) {
    const int cur = i & 1;
    if (i + 1 < NT) {
#pragma unroll
      for (int h = 0; h < 2; ++h)
        async_copy_b128(lds_off(&ldsW[cur ^ 1][(r0 + 32 * h) * 32 + c0]),
                        wsrc + (size_t)(32 * h) * DMODEL + (i + 1) * 32);
    }
    v16bf a = frag_f32(xrow, i * 32 + klo);
#pragma unroll
    for (int g = 0; g < 4; ++g) {
      v16bf b = frag_f32(&ldsW[cur][(g * 16 + l16) * 32], klo);
      acc[g] = wmma_bf16(a, b, acc[g]);
    }
    if (i + 1 < NT) { wait_async0(); __syncthreads(); }
  }

  if constexpr (!TRANS) {
#pragma unroll
    for (int g = 0; g < 4; ++g) {
      int ncol = ntile + g * 16 + l16;
      float bb = bias[ncol];
#pragma unroll
      for (int v = 0; v < 8; ++v) {
        int row = mtile + wave * 16 + v + 8 * lhalf;
        Out[(size_t)row * DMODEL + ncol] = (__bf16)(acc[g][v] + bb);
      }
    }
  } else {
    // Out = V^T: [NBATCH][DMODEL][SEQ]
    const int batch = mtile >> 11;                       // mtile / SEQ
    const int mm    = (mtile & (SEQ - 1)) + wave * 16 + 8 * lhalf;
#pragma unroll
    for (int g = 0; g < 4; ++g) {
      int ncol = ntile + g * 16 + l16;
      float bb = bias[ncol];
      bf16x8 pk;
#pragma unroll
      for (int v = 0; v < 8; ++v) pk[v] = (__bf16)(acc[g][v] + bb);
      *reinterpret_cast<bf16x8*>(
          Out + ((size_t)batch * DMODEL + ncol) * SEQ + mm) = pk;
    }
  }
}

// ---------------------------------------------------------------------------
// Kernel 2: scores = (Q*K^T)/32 per batch (M=N=2048,K=1024), f32 out.
// K tile (64 rows x 32 k, bf16) staged by TDM from wave 0, double-buffered.
// ---------------------------------------------------------------------------
__global__ __launch_bounds__(256)
void scores_kernel(const __bf16* __restrict__ Qb, const __bf16* __restrict__ Kb,
                   float* __restrict__ Sc) {
  __shared__ __bf16 ldsK[2][64 * 32];

  const int tid   = threadIdx.x;
  const int wave  = tid >> 5;
  const int lane  = tid & 31;
  const int l16   = lane & 15;
  const int lhalf = lane >> 4;
  const int klo   = lhalf * 8;

  const size_t bo = (size_t)blockIdx.z * SEQ * DMODEL;
  const __bf16* Q = Qb + bo;
  const __bf16* K = Kb + bo;
  float* S = Sc + (size_t)blockIdx.z * SEQ * SEQ;

  const int mtile = blockIdx.x * 128;   // query rows
  const int ntile = blockIdx.y * 64;    // key rows

  const __bf16* qrow  = Q + (size_t)(mtile + wave * 16 + l16) * DMODEL;
  const __bf16* ktile = K + (size_t)ntile * DMODEL;

  v8f acc[4] = {};
  const int NT = DMODEL / 32;

  if (wave == 0) {
    tdm_load_2d_bf16(lds_off(&ldsK[0][0]), ktile, 32, 64, DMODEL, DMODEL, SEQ);
    wait_tensor0();
  }
  __syncthreads();

  for (int i = 0; i < NT; ++i) {
    const int cur = i & 1;
    if (wave == 0 && i + 1 < NT)
      tdm_load_2d_bf16(lds_off(&ldsK[cur ^ 1][0]), ktile + (i + 1) * 32,
                       32, 64, DMODEL, DMODEL, SEQ);
    v16bf a = frag_bf16(qrow, i * 32 + klo);
#pragma unroll
    for (int g = 0; g < 4; ++g) {
      v16bf b = frag_bf16(&ldsK[cur][(g * 16 + l16) * 32], klo);
      acc[g] = wmma_bf16(a, b, acc[g]);
    }
    if (i + 1 < NT) {
      if (wave == 0) wait_tensor0();
      __syncthreads();
    }
  }

  const float scale = 0.03125f;   // 1/sqrt(1024)
#pragma unroll
  for (int g = 0; g < 4; ++g) {
#pragma unroll
    for (int v = 0; v < 8; ++v) {
      int row = mtile + wave * 16 + v + 8 * lhalf;
      int col = ntile + g * 16 + l16;
      S[(size_t)row * SEQ + col] = acc[g][v] * scale;
    }
  }
}

// ---------------------------------------------------------------------------
// Kernel 3: row softmax over 2048 keys; one wave32 per row; rewrites the row
// in place as normalized bf16 probabilities (row byte-stride stays 8192).
// ---------------------------------------------------------------------------
__global__ __launch_bounds__(256)
void softmax_kernel(float* __restrict__ Sc) {
  const int wave = threadIdx.x >> 5;
  const int lane = threadIdx.x & 31;
  float* rp = Sc + ((size_t)blockIdx.x * 8 + wave) * SEQ;

  float vals[64];
  float m = -1e30f;
#pragma unroll
  for (int i = 0; i < 64; ++i) {
    vals[i] = rp[i * 32 + lane];
    m = fmaxf(m, vals[i]);
  }
#pragma unroll
  for (int off = 16; off >= 1; off >>= 1)
    m = fmaxf(m, __shfl_xor(m, off, 32));

  float s = 0.0f;
#pragma unroll
  for (int i = 0; i < 64; ++i) {
    vals[i] = __expf(vals[i] - m);
    s += vals[i];
  }
#pragma unroll
  for (int off = 16; off >= 1; off >>= 1)
    s += __shfl_xor(s, off, 32);

  const float inv = 1.0f / s;
  __bf16* bp = reinterpret_cast<__bf16*>(rp);
#pragma unroll
  for (int i = 0; i < 64; ++i)
    bp[i * 32 + lane] = (__bf16)(vals[i] * inv);
}

// ---------------------------------------------------------------------------
// Kernel 4: O = P * V per batch (M=2048 q, N=1024 d, K=2048 keys), f32 out.
// V is pre-transposed ([D][S]); V^T tile staged by TDM, double-buffered.
// ---------------------------------------------------------------------------
__global__ __launch_bounds__(256)
void pv_kernel(const float* __restrict__ ScBase, const __bf16* __restrict__ VtB,
               float* __restrict__ OutB) {
  __shared__ __bf16 ldsV[2][64 * 32];

  const int tid   = threadIdx.x;
  const int wave  = tid >> 5;
  const int lane  = tid & 31;
  const int l16   = lane & 15;
  const int lhalf = lane >> 4;
  const int klo   = lhalf * 8;

  const char* scb = reinterpret_cast<const char*>(ScBase) +
                    (size_t)blockIdx.z * SEQ * SEQ * 4;
  const __bf16* Vt = VtB + (size_t)blockIdx.z * DMODEL * SEQ;
  float* Out       = OutB + (size_t)blockIdx.z * SEQ * DMODEL;

  const int qtile = blockIdx.x * 128;
  const int dtile = blockIdx.y * 64;

  const __bf16* prow = reinterpret_cast<const __bf16*>(
      scb + (size_t)(qtile + wave * 16 + l16) * (SEQ * 4));
  const __bf16* vtile = Vt + (size_t)dtile * SEQ;

  v8f acc[4] = {};
  const int NT = SEQ / 32;

  if (wave == 0) {
    tdm_load_2d_bf16(lds_off(&ldsV[0][0]), vtile, 32, 64, SEQ, SEQ, DMODEL);
    wait_tensor0();
  }
  __syncthreads();

  for (int i = 0; i < NT; ++i) {
    const int cur = i & 1;
    if (wave == 0 && i + 1 < NT)
      tdm_load_2d_bf16(lds_off(&ldsV[cur ^ 1][0]), vtile + (i + 1) * 32,
                       32, 64, SEQ, SEQ, DMODEL);
    v16bf a = frag_bf16(prow, i * 32 + klo);
#pragma unroll
    for (int g = 0; g < 4; ++g) {
      v16bf b = frag_bf16(&ldsV[cur][(g * 16 + l16) * 32], klo);
      acc[g] = wmma_bf16(a, b, acc[g]);
    }
    if (i + 1 < NT) {
      if (wave == 0) wait_tensor0();
      __syncthreads();
    }
  }

#pragma unroll
  for (int g = 0; g < 4; ++g) {
#pragma unroll
    for (int v = 0; v < 8; ++v) {
      int row = qtile + wave * 16 + v + 8 * lhalf;
      int col = dtile + g * 16 + l16;
      Out[(size_t)row * DMODEL + col] = acc[g][v];
    }
  }
}

// ---------------------------------------------------------------------------
extern "C" void kernel_launch(void* const* d_in, const int* in_sizes, int n_in,
                              void* d_out, int out_size, void* d_ws, size_t ws_size,
                              hipStream_t stream) {
  const float* x  = (const float*)d_in[0];
  const float* y  = (const float*)d_in[1];
  const float* z  = (const float*)d_in[2];
  const float* Wq = (const float*)d_in[3];
  const float* bq = (const float*)d_in[4];
  const float* Wk = (const float*)d_in[5];
  const float* bk = (const float*)d_in[6];
  const float* Wv = (const float*)d_in[7];
  const float* bv = (const float*)d_in[8];

  char* ws = (char*)d_ws;
  __bf16* Qb = (__bf16*)ws;
  __bf16* Kb = Qb + (size_t)MTOT * DMODEL;
  __bf16* Vt = Kb + (size_t)MTOT * DMODEL;     // [NBATCH][DMODEL][SEQ]
  float*  Sc = (float*)(ws + (size_t)3 * MTOT * DMODEL * 2);

  dim3 blk(256);
  dim3 gproj(MTOT / 128, DMODEL / 64);               // 64 x 16
  proj_kernel<false><<<gproj, blk, 0, stream>>>(x, Wq, bq, Qb);
  proj_kernel<false><<<gproj, blk, 0, stream>>>(y, Wk, bk, Kb);
  proj_kernel<true ><<<gproj, blk, 0, stream>>>(z, Wv, bv, Vt);

  dim3 gsc(SEQ / 128, SEQ / 64, NBATCH);             // 16 x 32 x 4
  scores_kernel<<<gsc, blk, 0, stream>>>(Qb, Kb, Sc);

  softmax_kernel<<<(NBATCH * SEQ) / 8, blk, 0, stream>>>(Sc);

  dim3 gpv(SEQ / 128, DMODEL / 64, NBATCH);          // 16 x 16 x 4
  pv_kernel<<<gpv, blk, 0, stream>>>(Sc, Vt, (float*)d_out);
}
